// SiameseNetwork_13460427506222
// MI455X (gfx1250) — matrix operand order repaired and checked by reading, not compiled, not run on hardware
//
#include <hip/hip_runtime.h>
#include <hip/hip_bf16.h>

#define B_    256
#define L_    256
#define D_    300
#define CO    230
#define COP   240
#define NB    512           // both inputs batched: 2*B
#define FCIN  2070
#define FCINP 2080
#define FCMID 690
#define FCMIDP 704
#define NCLS  53
#define NCLSP 64

typedef __attribute__((ext_vector_type(16))) __bf16 v16bf;
typedef __attribute__((ext_vector_type(8)))  float  v8f;
typedef __attribute__((ext_vector_type(4)))  unsigned int u32x4;
typedef __attribute__((ext_vector_type(2)))  unsigned int u32x2;
typedef __attribute__((ext_vector_type(8)))  int i32x8;
typedef __attribute__((ext_vector_type(4)))  int i32x4;

union FragQ { u32x4 q[2]; v16bf v; };   // two 16B global chunks
union FragD { u32x2 d[4]; v16bf v; };   // four 8B LDS chunks

__device__ __forceinline__ unsigned short f2bf(float f) {
  unsigned int u = __float_as_uint(f);
  unsigned int r = (u + 0x7FFFu + ((u >> 16) & 1u)) >> 16;
  return (unsigned short)r;
}

// ---------------- embedding gather + cast to bf16: xb[row, l, d] ----------------
__global__ void embed_cast_kernel(const int* __restrict__ in1, const int* __restrict__ in2,
                                  const float* __restrict__ tbl, unsigned short* __restrict__ xb) {
  long long idx = (long long)blockIdx.x * blockDim.x + threadIdx.x;
  if (idx >= (long long)NB * L_ * D_) return;
  int d  = (int)(idx % D_);
  int rl = (int)(idx / D_);
  int l  = rl % L_;
  int row = rl / L_;
  int tok = (row < B_) ? in1[row * L_ + l] : in2[(row - B_) * L_ + l];
  xb[idx] = f2bf(tbl[(long long)tok * D_ + d]);
}

// ------------- pack f32 weights (R,Kin) -> zero-padded bf16 (Rpad,Kp) -----------
__global__ void pack_w_kernel(const float* __restrict__ src, unsigned short* __restrict__ dst,
                              int R, int Kin, int Kp, long long total) {
  long long idx = (long long)blockIdx.x * blockDim.x + threadIdx.x;
  if (idx >= total) return;
  int k = (int)(idx % Kp);
  int r = (int)(idx / Kp);
  dst[idx] = (k < Kin && r < R) ? f2bf(src[(long long)r * Kin + k]) : (unsigned short)0;
}

// ---------------- fused conv (implicit GEMM) + bias + ReLU + piece-pool ----------------
// grid: (ceil(Tout/64), NB), block: 128 (4 waves). Wave w handles 16 output positions.
template<int K>
__global__ void conv_pool_kernel(const unsigned short* __restrict__ xb,
                                 const unsigned short* __restrict__ wb,
                                 const float* __restrict__ bias,
                                 const int* __restrict__ pos,
                                 float* __restrict__ pooled, int kidx) {
  constexpr int KD   = K * D_;
  constexpr int KDP  = ((KD + 31) / 32) * 32;
  constexpr int TOUT = L_ - K + 1;
  constexpr int MT   = 64;
  __shared__ __align__(16) unsigned short xs[68 * D_ + 64];

  const int b  = blockIdx.y;
  const int t0 = blockIdx.x * MT;

  const unsigned int nWords = (unsigned int)((MT + K - 1) * D_) / 2u;
  unsigned int* xsw = reinterpret_cast<unsigned int*>(xs);

#if __has_builtin(__builtin_amdgcn_tensor_load_to_lds) && __has_builtin(__builtin_amdgcn_s_wait_tensorcnt)
  // ---- Tensor Data Mover: one 2-D tile DMA (rows x 300 bf16) per workgroup ----
  if (threadIdx.x < 32) {        // TDM ignores EXEC; issue once from wave 0
    const int rows    = MT + K - 1;              // 66..68
    const int remRows = L_ - t0;                 // OOB rows beyond this are zero-filled
    unsigned long long ga =
        (unsigned long long)(uintptr_t)(xb + ((size_t)b * L_ + t0) * D_);
    u32x4 g0;
    g0.x = 1u;                                   // count=1 (valid user descriptor)
    g0.y = 0u;                                   // lds_addr: xs is the only LDS object -> 0
    g0.z = (unsigned int)(ga & 0xFFFFFFFFu);     // global_addr[31:0]
    g0.w = (unsigned int)((ga >> 32) & 0x01FFFFFFu) | (2u << 30);  // addr[56:32] | type=2
    i32x8 g1;
    g1[0] = (1 << 16);                                              // data_size=1 (2 bytes)
    g1[1] = (int)(((unsigned)D_ & 0xFFFFu) << 16);                  // tensor_dim0 lo16
    g1[2] = (int)((((unsigned)D_ >> 16) & 0xFFFFu) |
                  (((unsigned)remRows & 0xFFFFu) << 16));           // dim0 hi | dim1 lo
    g1[3] = (int)(((((unsigned)remRows) >> 16) & 0xFFFFu) |
                  (((unsigned)D_ & 0xFFFFu) << 16));                // dim1 hi | tile_dim0
    g1[4] = (int)((unsigned)rows & 0xFFFFu);                        // tile_dim1 (tile_dim2=0)
    g1[5] = D_;                                                     // tensor_dim0_stride lo32
    g1[6] = 0;                                                      // stride hi | dim1_stride lo
    g1[7] = 0;
    i32x4 z4 = {0, 0, 0, 0};
#if defined(__clang_major__) && __clang_major__ >= 23
    i32x8 z8 = {0, 0, 0, 0, 0, 0, 0, 0};
    __builtin_amdgcn_tensor_load_to_lds(g0, g1, z4, z4, z8, 0);
#else
    __builtin_amdgcn_tensor_load_to_lds(g0, g1, z4, z4, 0);
#endif
    __builtin_amdgcn_s_wait_tensorcnt(0);
  }
#else
  // ---- fallback: manual window fill, 4B words ----
  {
    const unsigned int* xw = reinterpret_cast<const unsigned int*>(xb);
    for (unsigned int wi = threadIdx.x; wi < nWords; wi += blockDim.x) {
      int e = (int)wi * 2;
      int r = e / D_;
      int c = e - r * D_;
      int ts = t0 + r; if (ts > L_ - 1) ts = L_ - 1;
      xsw[wi] = xw[(((size_t)b * L_ + ts) * D_ + c) >> 1];
    }
  }
#endif
  // zero slack beyond the window (padded weight region multiplies these)
  for (unsigned int wi = threadIdx.x; wi < 32u; wi += blockDim.x)
    xsw[nWords + wi] = 0u;
  __syncthreads();

  const int lane = threadIdx.x & 31;
  const int wave = threadIdx.x >> 5;
  const int m16  = lane & 15;
  const int g    = lane >> 4;
  const int tw   = wave * 16;

  int e1, e2;
  if (b < B_) { e1 = pos[b];               e2 = pos[B_ + b]; }
  else        { e1 = pos[2 * B_ + b - B_]; e2 = pos[3 * B_ + b - B_]; }
  if (e2 >= TOUT) e2 = e1 + 1;

  const unsigned short* xrow = xs + (tw + m16) * D_;

  for (int nt = 0; nt < COP / 16; ++nt) {
    const int n = nt * 16 + m16;
    const unsigned short* wrow = wb + (size_t)n * KDP;
    if (nt + 1 < COP / 16)
      __builtin_prefetch(wrow + (size_t)16 * KDP, 0, 1);   // global_prefetch next weight tile
    v8f acc = {};
    for (int kk = 0; kk < KDP; kk += 32) {
      FragD a;
      const u32x2* ap = reinterpret_cast<const u32x2*>(xrow + kk + g * 8);
      a.d[0] = ap[0]; a.d[1] = ap[1];
      const u32x2* ap2 = reinterpret_cast<const u32x2*>(xrow + kk + g * 8 + 16);
      a.d[2] = ap2[0]; a.d[3] = ap2[1];
      FragQ bf;
      const u32x4* bp = reinterpret_cast<const u32x4*>(wrow + kk + g * 8);
      bf.q[0] = bp[0]; bf.q[1] = bp[2];
      acc = __builtin_amdgcn_wmma_f32_16x16x32_bf16(false, a.v, false, bf.v,
                                                    (short)0, acc, false, false);
    }
    const float bn = (n < CO) ? bias[n] : 0.0f;
    float pm0 = -1.0f, pm1 = -1.0f, pm2 = -1.0f;
    for (int r = 0; r < 8; ++r) {
      int t = t0 + tw + r + 8 * g;
      if (t < TOUT) {
        float c = fmaxf(acc[r] + bn, 0.0f);
        if (t < e1)            pm0 = fmaxf(pm0, c);
        if (t >= e1 && t < e2) pm1 = fmaxf(pm1, c);
        if (t >= e2)           pm2 = fmaxf(pm2, c);
      }
    }
    if (n < CO) {
      float* base = pooled + (size_t)b * FCINP + kidx * (3 * CO) + n;
      if (pm0 >= 0.0f) atomicMax(reinterpret_cast<int*>(base),          __float_as_int(pm0));
      if (pm1 >= 0.0f) atomicMax(reinterpret_cast<int*>(base + CO),     __float_as_int(pm1));
      if (pm2 >= 0.0f) atomicMax(reinterpret_cast<int*>(base + 2 * CO), __float_as_int(pm2));
    }
  }
}

// ---------------- pooled f32 -> bf16 (pad cols already zero) ----------------
__global__ void cast_h0_kernel(const float* __restrict__ pooled, unsigned short* __restrict__ h0) {
  int idx = blockIdx.x * blockDim.x + threadIdx.x;
  if (idx < NB * FCINP) h0[idx] = f2bf(pooled[idx]);
}

// ---------------- generic bf16 WMMA FC: out = act(A @ W^T + b) ----------------
// grid: (Npad/16, NB/64), block 128 (4 waves).
__global__ void fc_kernel(const unsigned short* __restrict__ A,
                          const unsigned short* __restrict__ W,
                          const float* __restrict__ bias,
                          int Kp, int N, int Npad, int relu, int storeF32,
                          unsigned short* __restrict__ outBf,
                          float* __restrict__ outF32) {
  const int lane = threadIdx.x & 31;
  const int wave = threadIdx.x >> 5;
  const int m16  = lane & 15;
  const int g    = lane >> 4;
  const int m0   = blockIdx.y * 64 + wave * 16;
  const int n    = blockIdx.x * 16 + m16;

  const unsigned short* arow = A + (size_t)(m0 + m16) * Kp;
  const unsigned short* wrow = W + (size_t)n * Kp;
  v8f acc = {};
  for (int kk = 0; kk < Kp; kk += 32) {
    FragQ a, bf;
    a.q[0] = *reinterpret_cast<const u32x4*>(arow + kk + g * 8);
    a.q[1] = *reinterpret_cast<const u32x4*>(arow + kk + g * 8 + 16);
    const u32x4* bp = reinterpret_cast<const u32x4*>(wrow + kk + g * 8);
    bf.q[0] = bp[0]; bf.q[1] = bp[2];
    acc = __builtin_amdgcn_wmma_f32_16x16x32_bf16(false, a.v, false, bf.v,
                                                  (short)0, acc, false, false);
  }
  const float bn = (n < N) ? bias[n] : 0.0f;
  for (int r = 0; r < 8; ++r) {
    int m = m0 + r + 8 * g;
    float v = acc[r] + bn;
    if (relu) v = fmaxf(v, 0.0f);
    if (storeF32) {
      if (n < N) outF32[(size_t)m * N + n] = v;
    } else {
      outBf[(size_t)m * Npad + n] = (n < N) ? f2bf(v) : (unsigned short)0;
    }
  }
}

extern "C" void kernel_launch(void* const* d_in, const int* in_sizes, int n_in,
                              void* d_out, int out_size, void* d_ws, size_t ws_size,
                              hipStream_t stream) {
  const int*   in1 = (const int*)d_in[0];
  const int*   in2 = (const int*)d_in[1];
  const int*   pos = (const int*)d_in[2];
  const float* tbl = (const float*)d_in[3];
  const float* cw[3] = {(const float*)d_in[4], (const float*)d_in[6], (const float*)d_in[8]};
  const float* cb[3] = {(const float*)d_in[5], (const float*)d_in[7], (const float*)d_in[9]};
  const float* fw[4] = {(const float*)d_in[10], (const float*)d_in[12],
                        (const float*)d_in[14], (const float*)d_in[16]};
  const float* fb[4] = {(const float*)d_in[11], (const float*)d_in[13],
                        (const float*)d_in[15], (const float*)d_in[17]};

  const int KDP3 = 928, KDP4 = 1216, KDP5 = 1504;

  char* ws = (char*)d_ws;
  size_t off = 0;
  auto alloc = [&](size_t bytes) -> char* {
    char* p = ws + off;
    off += (bytes + 255) & ~(size_t)255;
    return p;
  };
  unsigned short* xb   = (unsigned short*)alloc((size_t)NB * L_ * D_ * 2);
  unsigned short* wb3  = (unsigned short*)alloc((size_t)COP * KDP3 * 2);
  unsigned short* wb4  = (unsigned short*)alloc((size_t)COP * KDP4 * 2);
  unsigned short* wb5  = (unsigned short*)alloc((size_t)COP * KDP5 * 2);
  unsigned short* fwp0 = (unsigned short*)alloc((size_t)FCINP * FCINP * 2);
  unsigned short* fwp1 = (unsigned short*)alloc((size_t)FCMIDP * FCINP * 2);
  unsigned short* fwp2 = (unsigned short*)alloc((size_t)FCMIDP * FCMIDP * 2);
  unsigned short* fwp3 = (unsigned short*)alloc((size_t)NCLSP * FCMIDP * 2);
  float*          pooled = (float*)alloc((size_t)NB * FCINP * 4);
  unsigned short* h0   = (unsigned short*)alloc((size_t)NB * FCINP * 2);
  unsigned short* h1   = (unsigned short*)alloc((size_t)NB * FCINP * 2);
  unsigned short* h2   = (unsigned short*)alloc((size_t)NB * FCMIDP * 2);
  unsigned short* h3   = (unsigned short*)alloc((size_t)NB * FCMIDP * 2);

  // pooled must start at 0 every call (atomic-max accumulation target)
  hipMemsetAsync(pooled, 0, (size_t)NB * FCINP * 4, stream);

  // embedding gather + cast
  {
    long long total = (long long)NB * L_ * D_;
    int blocks = (int)((total + 255) / 256);
    embed_cast_kernel<<<blocks, 256, 0, stream>>>(in1, in2, tbl, xb);
  }

  // weight packing (f32 -> padded bf16)
  auto pack = [&](const float* src, unsigned short* dst, int R, int Kin, int Rpad, int Kp) {
    long long total = (long long)Rpad * Kp;
    int blocks = (int)((total + 255) / 256);
    pack_w_kernel<<<blocks, 256, 0, stream>>>(src, dst, R, Kin, Kp, total);
  };
  pack(cw[0], wb3, CO, 3 * D_, COP, KDP3);
  pack(cw[1], wb4, CO, 4 * D_, COP, KDP4);
  pack(cw[2], wb5, CO, 5 * D_, COP, KDP5);
  pack(fw[0], fwp0, FCIN,  FCIN,  FCINP,  FCINP);
  pack(fw[1], fwp1, FCMID, FCIN,  FCMIDP, FCINP);
  pack(fw[2], fwp2, FCMID, FCMID, FCMIDP, FCMIDP);
  pack(fw[3], fwp3, NCLS,  FCMID, NCLSP,  FCMIDP);

  // convs + piecewise pooling (fused)
  conv_pool_kernel<3><<<dim3(4, NB), 128, 0, stream>>>(xb, wb3, cb[0], pos, pooled, 0);
  conv_pool_kernel<4><<<dim3(4, NB), 128, 0, stream>>>(xb, wb4, cb[1], pos, pooled, 1);
  conv_pool_kernel<5><<<dim3(4, NB), 128, 0, stream>>>(xb, wb5, cb[2], pos, pooled, 2);

  // pooled -> bf16
  cast_h0_kernel<<<(NB * FCINP + 255) / 256, 256, 0, stream>>>(pooled, h0);

  // MLP
  fc_kernel<<<dim3(FCINP / 16, NB / 64), 128, 0, stream>>>(
      h0, fwp0, fb[0], FCINP, FCIN, FCINP, 1, 0, h1, nullptr);
  fc_kernel<<<dim3(FCMIDP / 16, NB / 64), 128, 0, stream>>>(
      h1, fwp1, fb[1], FCINP, FCMID, FCMIDP, 1, 0, h2, nullptr);
  fc_kernel<<<dim3(FCMIDP / 16, NB / 64), 128, 0, stream>>>(
      h2, fwp2, fb[2], FCMIDP, FCMID, FCMIDP, 1, 0, h3, nullptr);
  fc_kernel<<<dim3(NCLSP / 16, NB / 64), 128, 0, stream>>>(
      h3, fwp3, fb[3], FCMIDP, NCLS, NCLSP, 0, 1, nullptr, (float*)d_out);
}
